// KernelAttentionEncoder_29918742184738
// MI455X (gfx1250) — matrix operand order repaired and checked by reading, not compiled, not run on hardware
//
#include <hip/hip_runtime.h>
#include <hip/hip_bf16.h>
#include <stdint.h>

typedef __attribute__((ext_vector_type(16))) _Float16 v16h;
typedef __attribute__((ext_vector_type(8)))  _Float16 v8h;
typedef __attribute__((ext_vector_type(8)))  float    v8f;

constexpr int Bc = 8;
constexpr int Nc = 2048;
constexpr int Dc = 128;
constexpr int Oc = 128;
constexpr int Hc = 3;
constexpr int VLDS_STRIDE = 40;   // halves; 80B row pitch -> conflict-free, 16B aligned
constexpr int NT = Nc / 32;       // 64 j-tiles

// ---------------------------------------------------------------------------
// Phase 1: V[h][b][n][o] = node_features[b,n,:] @ Wv[h,:,o]   (f32 -> f16)
// ---------------------------------------------------------------------------
__global__ __launch_bounds__(128)
void values_kernel(const float* __restrict__ nf, const float* __restrict__ Wv,
                   _Float16* __restrict__ V) {
    int idx = blockIdx.x;             // h*B*N + b*N + n
    int n  = idx % Nc;
    int hb = idx / Nc;
    int b  = hb % Bc;
    int h  = hb / Bc;
    int o  = threadIdx.x;
    const float* __restrict__ x = nf + (size_t)(b * Nc + n) * Dc;
    const float* __restrict__ w = Wv + (size_t)h * Dc * Oc + o;
    float s = 0.0f;
#pragma unroll 8
    for (int d = 0; d < Dc; ++d)
        s += x[d] * w[(size_t)d * Oc];
    V[((size_t)(h * Bc + b) * Nc + n) * Oc + o] = (_Float16)s;
}

// ---------------------------------------------------------------------------
// Phase 2: streaming attention with v_wmma_f32_16x16x32_f16.
// Fixed-max softmax (row max is statically 0: logits <= 0, diagonal == 0).
// Masking folded into additive bias terms (-3e9 -> exp underflows to 0):
// no compares, no selects, no clamp in the hot loop.
// Double-buffered LDS, register-staged pipeline, branchless steady state.
// Block = 256 threads = 8 waves; each wave owns a 16-row i-tile.
// ---------------------------------------------------------------------------
__global__ __launch_bounds__(256)
void attn_kernel(const float* __restrict__ coords,
                 const unsigned char* __restrict__ maskb,
                 const _Float16* __restrict__ V,
                 _Float16* __restrict__ Hout) {
    __shared__ __align__(16) _Float16 vlds[2][Oc * VLDS_STRIDE];  // 2 x 10240 B
    __shared__ float jx[2][32], jy[2][32], jz[2][32], jq[2][32];

    const int tile = blockIdx.x & 15;          // N/128 = 16 row tiles
    const int hb   = blockIdx.x >> 4;
    const int b    = hb & 7;
    const int h    = hb >> 3;
    const float inv_s2 = (h == 0) ? 1.0f : (h == 1 ? 0.25f : 0.0625f);
    const float t2     = 2.0f * inv_s2;

    const int tid  = threadIdx.x;
    const int lane = tid & 31;
    const int wv   = tid >> 5;
    const int mrow = lane & 15;                // A/B column-lane index
    const int hi   = lane >> 4;                // 0: k 0-7,16-23; 1: k 8-15,24-31
    const int i0   = tile * 128 + wv * 16;
    const int irow = i0 + mrow;

    const float cx = coords[(size_t)(b * Nc + irow) * 3 + 0];
    const float cy = coords[(size_t)(b * Nc + irow) * 3 + 1];
    const float cz = coords[(size_t)(b * Nc + irow) * 3 + 2];
    // row bias: -inv_s2*|ci|^2, or -3e9 if row masked (forces exp -> 0)
    float a = -inv_s2 * (cx * cx + cy * cy + cz * cz);
    if (maskb[b * Nc + irow]) a = -3.0e9f;

    float l = 0.0f;       // running row sum (row = lane&15); row max is 0
    v8f acc[8];
#pragma unroll
    for (int oc = 0; oc < 8; ++oc)
#pragma unroll
        for (int e = 0; e < 8; ++e) acc[oc][e] = 0.0f;

    const _Float16* __restrict__ vsrc = V + (size_t)(h * Bc + b) * Nc * Oc;

    // Each thread stages two v8h slots of the 32x128 V tile:
    //   slot0: row k0, halves [part*8, part*8+8);  slot1: row k0+16, same part.
    const int k0   = tid >> 4;                 // 0..15
    const int part = tid & 15;                 // 0..15

    v8h   r0, r1;
    float mx = 0.f, my = 0.f, mz = 0.f, mq = 0.f;

    auto load_regs = [&](int j0) {
        r0 = *(const v8h*)(vsrc + (size_t)(j0 + k0) * Oc + part * 8);
        r1 = *(const v8h*)(vsrc + (size_t)(j0 + k0 + 16) * Oc + part * 8);
        if (tid < 32) {
            int j = j0 + tid;
            mx = coords[(size_t)(b * Nc + j) * 3 + 0];
            my = coords[(size_t)(b * Nc + j) * 3 + 1];
            mz = coords[(size_t)(b * Nc + j) * 3 + 2];
            // col bias: -inv_s2*|cj|^2, or -3e9 if col masked
            mq = maskb[b * Nc + j] ? -3.0e9f
                                   : -inv_s2 * (mx * mx + my * my + mz * mz);
        }
    };
    auto store_regs = [&](int buf) {
#pragma unroll
        for (int e = 0; e < 8; ++e) {
            vlds[buf][(part * 8 + e) * VLDS_STRIDE + k0]      = r0[e];
            vlds[buf][(part * 8 + e) * VLDS_STRIDE + k0 + 16] = r1[e];
        }
        if (tid < 32) {
            jx[buf][tid] = mx; jy[buf][tid] = my; jz[buf][tid] = mz;
            jq[buf][tid] = mq;
        }
    };
    auto compute_tile = [&](int cur) {
        // --- P tile in WMMA A-layout: half c of lane <-> (M=lane&15, K=kk) ---
        float ts = 0.0f;
        v16h p;
#pragma unroll
        for (int c = 0; c < 16; ++c) {
            int kk = (c & 7) + (hi << 3) + ((c >= 8) ? 16 : 0);
            float dot = cx * jx[cur][kk];
            dot = __builtin_fmaf(cy, jy[cur][kk], dot);
            dot = __builtin_fmaf(cz, jz[cur][kk], dot);
            // logit = a + q_j + 2*inv_s2*(ci.cj) = -inv_s2*d2 (or <= -3e9 masked)
            float v  = __builtin_fmaf(dot, t2, a + jq[cur][kk]);
            float pc = __expf(v);                 // row max == 0; masked -> 0
            ts += pc;
            p[c] = (_Float16)pc;
        }
        ts += __shfl_xor(ts, 16, 32);
        l += ts;

        // --- P(16x32) x V(32x16) per 16-wide O chunk ---
#pragma unroll
        for (int oc = 0; oc < 8; ++oc) {
            int ocol = oc * 16 + mrow;
            const v8h* bp = (const v8h*)&vlds[cur][ocol * VLDS_STRIDE + (hi << 3)];
            v8h b0 = bp[0];
            v8h b1 = bp[2];
            v16h bm;
#pragma unroll
            for (int e = 0; e < 8; ++e) { bm[e] = b0[e]; bm[e + 8] = b1[e]; }
            acc[oc] = __builtin_amdgcn_wmma_f32_16x16x32_f16(
                false, p, false, bm, (short)0, acc[oc], false, false);
        }
    };

    load_regs(0);
    store_regs(0);
    __syncthreads();

#pragma unroll 1
    for (int jt = 0; jt < NT - 1; ++jt) {        // branchless steady state
        const int cur = jt & 1;
        load_regs((jt + 1) * 32);                // issue next-tile loads early
        compute_tile(cur);
        store_regs(cur ^ 1);                     // fill other buffer
        __syncthreads();
    }
    compute_tile((NT - 1) & 1);                  // peeled last tile

    // --- finalize: divide by row sums, store head output (f16) ---
    float linv = 1.0f / fmaxf(l, 1.0e-30f);
    float lrow[8];
#pragma unroll
    for (int r = 0; r < 8; ++r)
        lrow[r] = __shfl(linv, r + (hi << 3), 32);

#pragma unroll
    for (int oc = 0; oc < 8; ++oc) {
        int o = oc * 16 + mrow;
#pragma unroll
        for (int r = 0; r < 8; ++r) {
            int i = i0 + r + (hi << 3);
            Hout[((size_t)(b * Nc + i) * Hc + h) * Oc + o] =
                (_Float16)(acc[oc][r] * lrow[r]);
        }
    }
}

// ---------------------------------------------------------------------------
// Phase 3: out[b,n,:] = head_out[b,n,:] @ Wo + bo, zero masked rows.
// ---------------------------------------------------------------------------
__global__ __launch_bounds__(128)
void outproj_kernel(const _Float16* __restrict__ Hout,
                    const float* __restrict__ Wo,
                    const float* __restrict__ bo,
                    const unsigned char* __restrict__ maskb,
                    float* __restrict__ out) {
    int bn = blockIdx.x;
    int o  = threadIdx.x;
    const _Float16* __restrict__ hrow = Hout + (size_t)bn * (Hc * Oc);
    float s = bo[o];
#pragma unroll 6
    for (int kc = 0; kc < (Hc * Oc) / 8; ++kc) {
        v8h hv = *(const v8h*)(hrow + kc * 8);
#pragma unroll
        for (int e = 0; e < 8; ++e)
            s += (float)hv[e] * Wo[(size_t)(kc * 8 + e) * Oc + o];
    }
    if (maskb[bn]) s = 0.0f;
    out[(size_t)bn * Oc + o] = s;
}

// ---------------------------------------------------------------------------
extern "C" void kernel_launch(void* const* d_in, const int* in_sizes, int n_in,
                              void* d_out, int out_size, void* d_ws, size_t ws_size,
                              hipStream_t stream) {
    const float*         nf     = (const float*)d_in[0];
    const float*         coords = (const float*)d_in[1];
    const unsigned char* maskb  = (const unsigned char*)d_in[2];  // bool [B,N]
    const float*         Wv     = (const float*)d_in[3];
    const float*         Wo     = (const float*)d_in[4];
    const float*         bo     = (const float*)d_in[5];
    float*               out    = (float*)d_out;

    _Float16* V    = (_Float16*)d_ws;                                   // 12.6 MB
    _Float16* Hout = V + (size_t)Hc * Bc * Nc * Oc;                     // 12.6 MB

    values_kernel<<<Hc * Bc * Nc, 128, 0, stream>>>(nf, Wv, V);
    attn_kernel<<<Bc * Hc * (Nc / 128), 256, 0, stream>>>(coords, maskb, V, Hout);
    outproj_kernel<<<Bc * Nc, 128, 0, stream>>>(Hout, Wo, bo, maskb, out);
}
